// MOE_84499186582152
// MI455X (gfx1250) — compile-verified, matrix operand order checked by reading, not run on hardware
//
#include <hip/hip_runtime.h>

#define D_MODEL 1024
#define H_FFN   4096
#define NEXP    12
#define NTOK    2048
#define TM      16
#define TILES   (NTOK / TM)   // 128 token tiles
#define HC      128           // H chunk per iteration
#define NCHUNK  (H_FFN / HC)  // 32
#define KSUB    128           // K sub-tile staged per TDM transfer
#define NSTAGE  (D_MODEL / KSUB)   // 8 up-proj stages per chunk

// dynamic LDS layout (bytes)
#define XS_OFF  0                      // 16 x 1024 bf16 = 32768
#define WT_OFF  32768                  // 2 x 128 x 128 bf16 = 65536 (ping-pong)
#define HS_OFF  (32768 + 65536)        // 16 x 128 bf16 = 4096
#define TOK_OFF (HS_OFF + 4096)        // 16 ints
#define CW_OFF  (TOK_OFF + 64)         // 16 floats
#define SMEM_BYTES (CW_OFF + 64)       // 102528

typedef __attribute__((ext_vector_type(16))) __bf16 v16bf;
typedef __attribute__((ext_vector_type(8)))  __bf16 v8bf;
typedef __attribute__((ext_vector_type(8)))  float  v8f;
typedef __attribute__((ext_vector_type(4)))  unsigned int u32x4;
typedef __attribute__((ext_vector_type(8)))  int i32x8;
typedef __attribute__((ext_vector_type(4)))  int i32x4;

static __device__ __forceinline__ unsigned short f2bf_bits(float f) {
    unsigned u = __builtin_bit_cast(unsigned, f);
    u += 0x7FFFu + ((u >> 16) & 1u);      // round-to-nearest-even
    return (unsigned short)(u >> 16);
}

// two contiguous 8x bf16 runs -> one 16-element fragment (per CDNA5 16-bit layout)
static __device__ __forceinline__ v16bf frag16(const unsigned short* p0,
                                               const unsigned short* p1) {
    v8bf lo = *(const v8bf*)p0;
    v8bf hi = *(const v8bf*)p1;
    return __builtin_shufflevector(lo, hi, 0, 1, 2, 3, 4, 5, 6, 7,
                                   8, 9, 10, 11, 12, 13, 14, 15);
}

// ---- TDM: 2D tile (tile_d0 x tile_d1 elements of 2 bytes) global -> LDS -----
static __device__ __forceinline__ void tdm_load_2d(unsigned lds_off, const void* gptr,
                                                   unsigned tile_d0, unsigned tile_d1,
                                                   unsigned long long row_stride,
                                                   unsigned tensor_d0, unsigned tensor_d1) {
    unsigned long long ga = (unsigned long long)(uintptr_t)gptr;
    u32x4 g0;
    g0[0] = 1u;                                         // count=1, no gather
    g0[1] = lds_off;                                    // LDS byte address
    g0[2] = (unsigned)ga;                               // global addr [31:0]
    g0[3] = (unsigned)((ga >> 32) & 0x1FFFFFFu) | (2u << 30);  // addr[56:32] | type=2
    i32x8 g1;
    g1[0] = (int)(1u << 16);                            // data_size = 2 bytes
    g1[1] = (int)((tensor_d0 & 0xFFFFu) << 16);         // tensor_dim0 lo16
    g1[2] = (int)((tensor_d0 >> 16) | ((tensor_d1 & 0xFFFFu) << 16));
    g1[3] = (int)((tensor_d1 >> 16) | (tile_d0 << 16)); // tile_dim0
    g1[4] = (int)(tile_d1 & 0xFFFFu);                   // tile_dim1, tile_dim2=0
    g1[5] = (int)(unsigned)(row_stride & 0xFFFFFFFFull);        // dim0 stride lo32
    g1[6] = (int)(unsigned)((row_stride >> 32) & 0xFFFFull);    // dim0 stride hi16
    g1[7] = 0;
    i32x4 z4 = { 0, 0, 0, 0 };
    i32x8 z8 = { 0, 0, 0, 0, 0, 0, 0, 0 };
    __builtin_amdgcn_tensor_load_to_lds(g0, g1, z4, z4, z8, 0);
}

// ---- TDM gather: 16 rows (16-bit indices) of a 2D tensor -> LDS -------------
static __device__ __forceinline__ void tdm_gather_rows(unsigned lds_off, const void* gptr,
                                                       unsigned row_len, unsigned nrows_tensor,
                                                       const unsigned* idx16 /*16 vals*/) {
    unsigned long long ga = (unsigned long long)(uintptr_t)gptr;
    u32x4 g0;
    g0[0] = 1u | (1u << 31);                            // count=1, gather_mode, 16-bit idx
    g0[1] = lds_off;
    g0[2] = (unsigned)ga;
    g0[3] = (unsigned)((ga >> 32) & 0x1FFFFFFu) | (2u << 30);
    i32x8 g1;
    g1[0] = (int)(1u << 16);                            // data_size = 2 bytes
    g1[1] = (int)((row_len & 0xFFFFu) << 16);           // tensor_dim0
    g1[2] = (int)((row_len >> 16) | ((nrows_tensor & 0xFFFFu) << 16));
    g1[3] = (int)((nrows_tensor >> 16) | (row_len << 16)); // tile_dim0 = row_len
    g1[4] = 16;                                         // tile_dim1 = #indices
    g1[5] = (int)row_len;                               // dim0 stride
    g1[6] = 0;
    g1[7] = 0;
    i32x4 g2, g3;
#pragma unroll
    for (int i = 0; i < 4; ++i)
        g2[i] = (int)(idx16[2 * i] | (idx16[2 * i + 1] << 16));
#pragma unroll
    for (int i = 0; i < 4; ++i)
        g3[i] = (int)(idx16[8 + 2 * i] | (idx16[8 + 2 * i + 1] << 16));
    i32x8 z8 = { 0, 0, 0, 0, 0, 0, 0, 0 };
    __builtin_amdgcn_tensor_load_to_lds(g0, g1, g2, g3, z8, 0);
}

// ---------------------------------------------------------------- zero / init
__global__ void moe_zero(float* __restrict__ out, int* __restrict__ counts, int n) {
    int i = blockIdx.x * blockDim.x + threadIdx.x;
    if (i < n) out[i] = 0.0f;
    if (i < NEXP) counts[i] = 0;
}

// ---------------------------------------------------------------- x -> bf16
__global__ void moe_conv_x(const float* __restrict__ x, unsigned int* __restrict__ xbf) {
    int i = blockIdx.x * blockDim.x + threadIdx.x;   // one u32 = 2 bf16
    if (i >= NTOK * D_MODEL / 2) return;
    float a = x[2 * i], b = x[2 * i + 1];
    xbf[i] = (unsigned)f2bf_bits(a) | ((unsigned)f2bf_bits(b) << 16);
}

// --------------------------------------------- fp32 [m][R][C] -> bf16 [m][C][R]
__global__ void moe_transpose_bf16(const float* __restrict__ src,
                                   unsigned short* __restrict__ dst, int R, int C) {
    __shared__ float tile[32][33];
    int tilesC = C >> 5, tilesR = R >> 5;
    int b  = blockIdx.x;
    int tc = b % tilesC; b /= tilesC;
    int tr = b % tilesR; b /= tilesR;
    const float* s = src + (size_t)b * R * C;
    unsigned short* d = dst + (size_t)b * R * C;
    int tx = threadIdx.x & 31, ty = threadIdx.x >> 5;   // 32x8
#pragma unroll
    for (int r = 0; r < 4; ++r) {
        int row = tr * 32 + ty + r * 8;
        tile[ty + r * 8][tx] = s[(size_t)row * C + tc * 32 + tx];
    }
    __syncthreads();
#pragma unroll
    for (int r = 0; r < 4; ++r) {
        int orow = tc * 32 + ty + r * 8;                // new row = old column
        d[(size_t)orow * R + tr * 32 + tx] = f2bf_bits(tile[tx][ty + r * 8]);
    }
}

// ---------------------------------------------------------------- gating
__global__ void moe_gate(const float* __restrict__ x, const float* __restrict__ Wg,
                         int* __restrict__ counts, int* __restrict__ toks,
                         float* __restrict__ tw) {
    const int wave  = threadIdx.x >> 5;
    const int lane  = threadIdx.x & 31;
    const int token = blockIdx.x * 4 + wave;
    if (token >= NTOK) return;

    float acc[NEXP];
#pragma unroll
    for (int e = 0; e < NEXP; ++e) acc[e] = 0.0f;

    const float* xp = x + (size_t)token * D_MODEL;
    for (int d = lane; d < D_MODEL; d += 32) {
        float xv = xp[d];
        const float* wg = Wg + (size_t)d * NEXP;
#pragma unroll
        for (int e = 0; e < NEXP; ++e) acc[e] += xv * wg[e];
    }
#pragma unroll
    for (int e = 0; e < NEXP; ++e) {
#pragma unroll
        for (int off = 16; off > 0; off >>= 1)
            acc[e] += __shfl_xor(acc[e], off, 32);
    }
    if (lane == 0) {
        int   i0 = -1, i1 = -1, i2 = -1;
        float v0 = -1e30f, v1 = -1e30f, v2 = -1e30f;
#pragma unroll
        for (int e = 0; e < NEXP; ++e) {
            float v = acc[e];
            if (v > v0)      { v2 = v1; i2 = i1; v1 = v0; i1 = i0; v0 = v; i0 = e; }
            else if (v > v1) { v2 = v1; i2 = i1; v1 = v;  i1 = e; }
            else if (v > v2) { v2 = v;  i2 = e; }
        }
        float e0 = 1.0f, e1 = expf(v1 - v0), e2 = expf(v2 - v0);
        float s  = e0 + e1 + e2;
        int   idx[3] = { i0, i1, i2 };
        float w[3]   = { e0 / s, e1 / s, e2 / s };
#pragma unroll
        for (int k = 0; k < 3; ++k) {
            int e    = idx[k];
            int slot = atomicAdd(&counts[e], 1);
            toks[e * NTOK + slot] = token;
            tw  [e * NTOK + slot] = w[k];
        }
    }
}

// ---------------------------------------------------------------- fused expert FFN
__launch_bounds__(256, 1)
__global__ void moe_expert_gemm(const unsigned short* __restrict__ xbf,   // [NTOK][D] bf16
                                const unsigned short* __restrict__ WupT,  // [E][H][D] bf16
                                const float* __restrict__ bup,
                                const unsigned short* __restrict__ WdnT,  // [E][D][H] bf16
                                const float* __restrict__ bdn,
                                const int*   __restrict__ counts,
                                const int*   __restrict__ toks,
                                const float* __restrict__ tw,
                                float* __restrict__ out) {
    extern __shared__ unsigned char smem[];
    unsigned short (*Xs)[D_MODEL]     = (unsigned short(*)[D_MODEL])(smem + XS_OFF);
    unsigned short (*Wt)[128][KSUB]   = (unsigned short(*)[128][KSUB])(smem + WT_OFF);
    unsigned short (*Hs)[HC]          = (unsigned short(*)[HC])(smem + HS_OFF);
    int*   tok_s = (int*)(smem + TOK_OFF);
    float* cw_s  = (float*)(smem + CW_OFF);

    const int e    = blockIdx.x / TILES;
    const int tile = blockIdx.x % TILES;
    const int cnt  = counts[e];
    if (tile * TM >= cnt) return;

    const int tid   = threadIdx.x;
    const int wave  = tid >> 5;     // 0..7
    const int lane  = tid & 31;
    const int half8 = (lane >> 4) << 3;   // 0 or 8
    const int lq    = lane & 15;
    const int nloc  = wave * 16 + lq;     // this lane's row in the staged weight tile

    if (tid < TM) {
        int slot = tile * TM + tid;
        tok_s[tid] = (slot < cnt) ? toks[e * NTOK + slot] : -1;
        cw_s [tid] = (slot < cnt) ? tw  [e * NTOK + slot] : 0.0f;
    }
    __syncthreads();

    const unsigned smem_base = (unsigned)(uintptr_t)&smem[0];
    const unsigned xs_off = smem_base + XS_OFF;
    const unsigned wt_off0 = smem_base + WT_OFF;
    const unsigned wt_off1 = smem_base + WT_OFF + 128 * KSUB * 2;

    // TDM gather of the 16 routed token rows of bf16 x into Xs.
    // No wait: TENSORcnt completes in order, so the first stage's wait covers it.
    if (wave == 0) {
        unsigned idx[16];
#pragma unroll
        for (int i = 0; i < 16; ++i) {
            int t = tok_s[i];
            idx[i] = (t < 0) ? 0xFFFFu : (unsigned)t;   // OOB rows read as zero
        }
        tdm_gather_rows(xs_off, xbf, D_MODEL, NTOK, idx);
    }

    const unsigned short* WupT_e = WupT + (size_t)e * H_FFN * D_MODEL;
    const unsigned short* WdnT_e = WdnT + (size_t)e * D_MODEL * H_FFN;

    const v8f vzero = { 0.f, 0.f, 0.f, 0.f, 0.f, 0.f, 0.f, 0.f };
    v8f yacc[8];
#pragma unroll
    for (int t = 0; t < 8; ++t) yacc[t] = vzero;

    for (int ch = 0; ch < NCHUNK; ++ch) {
        const int hbase = ch * HC;

        // ---------------- up-proj: wave owns h columns [hbase + wave*16, +16)
        if (wave == 0)   // prologue: stage 0 into buffer 0
            tdm_load_2d(wt_off0, WupT_e + (size_t)hbase * D_MODEL,
                        KSUB, 128, D_MODEL, D_MODEL, H_FFN);
        v8f hc = vzero;
        for (int s = 0; s < NSTAGE; ++s) {
            const int ks = s * KSUB;
            if (wave == 0) {
                if (s + 1 < NSTAGE) {  // issue next stage into the other buffer
                    tdm_load_2d((s & 1) ? wt_off0 : wt_off1,
                                WupT_e + (size_t)hbase * D_MODEL + ks + KSUB,
                                KSUB, 128, D_MODEL, D_MODEL, H_FFN);
                    __builtin_amdgcn_s_wait_tensorcnt(1);  // stage s complete
                } else {
                    __builtin_amdgcn_s_wait_tensorcnt(0);
                }
            }
            __syncthreads();
            const unsigned short* wrow = &Wt[s & 1][nloc][0];
#pragma unroll
            for (int kbl = 0; kbl < KSUB; kbl += 32) {
                const unsigned short* ar = &Xs[lq][ks + kbl + half8];
                const unsigned short* br = wrow + kbl + half8;
                v16bf a = frag16(ar, ar + 16);
                v16bf b = frag16(br, br + 16);
                hc = __builtin_amdgcn_wmma_f32_16x16x32_bf16(false, a, false, b,
                                                             (short)0, hc, false, false);
            }
            __syncthreads();   // all reads done before this buffer is re-filled
        }

        // prologue for down-proj overlapped with the activation epilogue
        if (wave == 0)
            tdm_load_2d(wt_off0, WdnT_e + hbase,
                        KSUB, 128, H_FFN, H_FFN, D_MODEL);

        // bias + exact GELU -> Hs (bf16)
        {
            const int hn = hbase + nloc;
            float bia = bup[e * H_FFN + hn];
#pragma unroll
            for (int r = 0; r < 8; ++r) {
                float v = hc[r] + bia;
                float g = 0.5f * v * (1.0f + erff(v * 0.70710678118654752440f));
                Hs[r + half8][nloc] = f2bf_bits(g);
            }
        }
        __syncthreads();

        // ---------------- down-proj: 8 d-blocks of 128, wave owns 16 rows each
        for (int sd = 0; sd < 8; ++sd) {
            const int db = sd * 128;
            if (wave == 0) {
                if (sd + 1 < 8) {
                    tdm_load_2d((sd & 1) ? wt_off0 : wt_off1,
                                WdnT_e + (size_t)(db + 128) * H_FFN + hbase,
                                KSUB, 128, H_FFN, H_FFN, D_MODEL);
                    __builtin_amdgcn_s_wait_tensorcnt(1);
                } else {
                    __builtin_amdgcn_s_wait_tensorcnt(0);
                }
            }
            __syncthreads();
            const unsigned short* wrow = &Wt[sd & 1][nloc][0];
#pragma unroll
            for (int kbl = 0; kbl < HC; kbl += 32) {
                const unsigned short* ar = &Hs[lq][kbl + half8];
                const unsigned short* br = wrow + kbl + half8;
                v16bf a = frag16(ar, ar + 16);
                v16bf b = frag16(br, br + 16);
                yacc[sd] = __builtin_amdgcn_wmma_f32_16x16x32_bf16(false, a, false, b,
                                                                   (short)0, yacc[sd], false, false);
            }
            __syncthreads();
        }
    }

    // epilogue: out[token] += cw * (y + bdown)
#pragma unroll
    for (int t = 0; t < 8; ++t) {
        const int dn  = t * 128 + nloc;
        const float bia = bdn[e * D_MODEL + dn];
#pragma unroll
        for (int r = 0; r < 8; ++r) {
            int row = r + half8;
            int tok = tok_s[row];
            if (tok >= 0) {
                float val = cw_s[row] * (yacc[t][r] + bia);
                atomicAdd(&out[(size_t)tok * D_MODEL + dn], val);
            }
        }
    }
}

// ---------------------------------------------------------------- launcher
extern "C" void kernel_launch(void* const* d_in, const int* in_sizes, int n_in,
                              void* d_out, int out_size, void* d_ws, size_t ws_size,
                              hipStream_t stream) {
    const float* x    = (const float*)d_in[0];
    const float* Wg   = (const float*)d_in[1];
    const float* Wup  = (const float*)d_in[2];
    const float* bup  = (const float*)d_in[3];
    const float* Wdn  = (const float*)d_in[4];
    const float* bdn  = (const float*)d_in[5];
    float* out = (float*)d_out;

    char* ws = (char*)d_ws;
    int*   counts = (int*)ws;
    int*   toks   = (int*)(ws + 256);
    float* tw     = (float*)(ws + 256 + (size_t)NEXP * NTOK * sizeof(int));
    unsigned short* xbf  = (unsigned short*)(ws + (1ull << 22));    //   4 MB
    unsigned short* WupT = (unsigned short*)(ws + (16ull << 20));   //  16 MB
    unsigned short* WdnT = (unsigned short*)(ws + (16ull << 20)
                               + (size_t)NEXP * D_MODEL * H_FFN * 2);

    (void)hipFuncSetAttribute((const void*)moe_expert_gemm,
                              hipFuncAttributeMaxDynamicSharedMemorySize, SMEM_BYTES);

    const int n = NTOK * D_MODEL;
    moe_zero<<<(n + 255) / 256, 256, 0, stream>>>(out, counts, n);
    moe_conv_x<<<(NTOK * D_MODEL / 2 + 255) / 256, 256, 0, stream>>>(x, (unsigned int*)xbf);
    moe_transpose_bf16<<<NEXP * (D_MODEL / 32) * (H_FFN / 32), 256, 0, stream>>>(
        Wup, WupT, D_MODEL, H_FFN);
    moe_transpose_bf16<<<NEXP * (H_FFN / 32) * (D_MODEL / 32), 256, 0, stream>>>(
        Wdn, WdnT, H_FFN, D_MODEL);
    moe_gate<<<(NTOK + 3) / 4, 128, 0, stream>>>(x, Wg, counts, toks, tw);
    moe_expert_gemm<<<NEXP * TILES, 256, SMEM_BYTES, stream>>>(xbf, WupT, bup, WdnT, bdn,
                                                               counts, toks, tw, out);
}
